// LeechQuantizerUnitVol_70274254897527
// MI455X (gfx1250) — matrix operand order repaired and checked by reading, not compiled, not run on hardware
//
#include <hip/hip_runtime.h>
#include <stdint.h>

typedef __attribute__((ext_vector_type(16))) _Float16 v16h;
typedef __attribute__((ext_vector_type(8)))  float    v8f;
typedef __attribute__((__vector_size__(16))) int      vi4;   // matches async-LDS builtin param

#define DIM      24
#define BATCH    1024
#define KCOSET   8192
#define NBROW    64      // BATCH/16
#define NKTILE   512     // KCOSET/16
#define P96      96
#define INV_A    2.8284271247461903f   // sqrt(8)
#define A_SCALE  0.3535533905932738f   // 1/sqrt(8)

// ---- workspace layout (bytes) --------------------------------------------
#define WS_CELLS  0u                        // 1024 * u64   packed {D_bits,k}, atomicMin
#define WS_MK     8192u                     // 8192 * u64   packed (C mod 4) codes + Q-parity bit63
#define WS_ETAB   (WS_MK + 65536u)          // 1024*96 f32  |d| tables (emax correction)
#define WS_AFRAG  (WS_ETAB + 393216u)       // 64*3*3*32*16 f16  pre-swizzled WMMA A fragments
#define WS_BFRAG  (WS_AFRAG + 589824u)      // 512*3*32*16  f16  pre-swizzled one-hot B fragments
// total ~2.51 MB

#ifndef __has_builtin
#define __has_builtin(x) 0
#endif
#if __has_builtin(__builtin_amdgcn_global_load_async_to_lds_b128)
#define USE_ASYNC_LDS 1
#else
#define USE_ASYNC_LDS 0
#endif

// ---- kernel 1: per-b lookup tables + pre-swizzled A fragments ------------
__global__ void leech_prep_tables(const float* __restrict__ x_in,
                                  uint8_t* __restrict__ ws)
{
    int tid  = blockIdx.x * blockDim.x + threadIdx.x;   // 0..2047
    int brow = tid >> 5;
    int lane = tid & 31;
    int bl   = lane & 15;
    int b    = brow * 16 + bl;

    float e2[P96];
    float ea[P96];
    int   pib[P96];
    const float* xr = x_in + b * DIM;
    #pragma unroll
    for (int i = 0; i < DIM; ++i) {
        float xs = xr[i] * INV_A;              // x = x_in / A
        #pragma unroll
        for (int m = 0; m < 4; ++m) {
            float t = (xs - (float)m) * 0.25f; // t depends on C only via C mod 4
            float g = rintf(t);
            float d = t - g;
            int   p = i * 4 + m;
            e2[p]  = d * d;
            ea[p]  = fabsf(d);
            pib[p] = ((int)g) & 1;             // parity of rint
        }
    }
    if (lane < 16) {                           // one writer per b
        float* et = (float*)(ws + WS_ETAB) + b * P96;
        for (int p = 0; p < P96; ++p) et[p] = ea[p];
    }
    // A fragments: 16x32 f16 layout. lane<16 -> K 0..7 / 16..23 ; lane>=16 -> K 8..15 / 24..31
    _Float16* af = (_Float16*)(ws + WS_AFRAG);
    #pragma unroll
    for (int s = 0; s < 3; ++s) {
        int fh = (((brow * 3 + s) * 3 + 0) * 32 + lane) * 16;
        int fl = (((brow * 3 + s) * 3 + 1) * 32 + lane) * 16;
        int fp = (((brow * 3 + s) * 3 + 2) * 32 + lane) * 16;
        #pragma unroll
        for (int j = 0; j < 16; ++j) {
            int K = ((lane >= 16) ? 8 : 0) + ((j < 8) ? j : (j + 8));
            int p = s * 32 + K;
            _Float16 hi = (_Float16)e2[p];
            float    lo = e2[p] - (float)hi;   // exact split: hi + lo == e2 (f32)
            af[fh + j] = hi;
            af[fl + j] = (_Float16)lo;
            af[fp + j] = (_Float16)(float)pib[p];
        }
    }
}

// ---- kernel 2: per-k packed (C mod 4) code + integer-part parity ---------
__global__ void leech_prep_cosets(const int* __restrict__ C_rep,
                                  uint8_t* __restrict__ ws)
{
    int k = blockIdx.x * blockDim.x + threadIdx.x;      // 0..8191
    const int* cr = C_rep + k * DIM;
    unsigned long long mk = 0ull;
    int qs = 0;
    #pragma unroll
    for (int i = 0; i < DIM; ++i) {
        int c = cr[i];
        int m = c & 3;                 // floor-mod for power of 2
        int q = (c - m) >> 2;          // exact integer part
        mk |= (unsigned long long)(unsigned)m << (2 * i);
        qs += q;
    }
    mk |= (unsigned long long)(qs & 1) << 63;
    ((unsigned long long*)(ws + WS_MK))[k] = mk;
    if (k < BATCH)                     // init argmin cells every call (deterministic)
        ((unsigned long long*)(ws + WS_CELLS))[k] = ~0ull;
}

// ---- kernel 2b: pre-swizzled one-hot B fragments (once per k, not per tile)
__global__ void leech_prep_bfrag(const int* __restrict__ C_rep,
                                 uint8_t* __restrict__ ws)
{
    int tid   = blockIdx.x * blockDim.x + threadIdx.x;  // 0..16383
    int ktile = tid >> 5;
    int lane  = tid & 31;
    int k     = ktile * 16 + (lane & 15);               // B layout: N = lane & 15
    int kbase = (lane >= 16) ? 8 : 0;                   // lanes 0-15: K 0-7/16-23, 16-31: K 8-15/24-31
    const int* cr = C_rep + k * DIM;
    unsigned* bf = (unsigned*)(ws + WS_BFRAG);
    #pragma unroll
    for (int s = 0; s < 3; ++s) {
        #pragma unroll
        for (int jj = 0; jj < 8; ++jj) {
            int p0 = s * 32 + kbase + 2 * jj + ((jj >= 4) ? 8 : 0);
            int p1 = p0 + 1;
            unsigned lo = ((cr[p0 >> 2] & 3) == (p0 & 3)) ? 0x3C00u : 0u;
            unsigned hi = ((cr[p1 >> 2] & 3) == (p1 & 3)) ? 0x3C00u : 0u;
            bf[(((ktile * 3 + s) * 32) + lane) * 8 + jj] = lo | (hi << 16);
        }
    }
}

// ---- kernel 3: WMMA lookup-GEMM + parity correction + argmin -------------
__global__ void __launch_bounds__(128)
leech_gemm(uint8_t* __restrict__ ws)
{
    __shared__ float eT[16 * P96];     // |d| tables for this brow's 16 batch rows (6KB)
    int brow  = blockIdx.x;
    int wid   = threadIdx.x >> 5;
    int lane  = threadIdx.x & 31;
    int ktile = blockIdx.y * 4 + wid;  // 128 blocks * 4 waves = 512 ktiles

    {   // stage |d| tables: 48B per thread
        const float* src = (const float*)(ws + WS_ETAB) + brow * 16 * P96 + threadIdx.x * 12;
        float* dst = eT + threadIdx.x * 12;
#if USE_ASYNC_LDS
        __builtin_amdgcn_global_load_async_to_lds_b128((vi4*)(src + 0), (vi4*)(dst + 0), 0, 0);
        __builtin_amdgcn_global_load_async_to_lds_b128((vi4*)(src + 4), (vi4*)(dst + 4), 0, 0);
        __builtin_amdgcn_global_load_async_to_lds_b128((vi4*)(src + 8), (vi4*)(dst + 8), 0, 0);
#if __has_builtin(__builtin_amdgcn_s_wait_asynccnt)
        __builtin_amdgcn_s_wait_asynccnt(0);
#else
        asm volatile("s_wait_asynccnt 0" ::: "memory");
#endif
#else
        const float4* s4 = (const float4*)src;
        float4* d4 = (float4*)dst;
        d4[0] = s4[0]; d4[1] = s4[1]; d4[2] = s4[2];
#endif
    }
    __syncthreads();

    const v16h* af = (const v16h*)(ws + WS_AFRAG);
    const v16h* bf = (const v16h*)(ws + WS_BFRAG);
    v16h Ahi[3], Alo[3], Api[3], Bv[3];
    #pragma unroll
    for (int s = 0; s < 3; ++s) {
        Ahi[s] = af[((brow * 3 + s) * 3 + 0) * 32 + lane];
        Alo[s] = af[((brow * 3 + s) * 3 + 1) * 32 + lane];
        Api[s] = af[((brow * 3 + s) * 3 + 2) * 32 + lane];
        Bv[s]  = bf[(ktile * 3 + s) * 32 + lane];
    }

    int kk = ktile * 16 + (lane & 15);                 // C layout: N = lane & 15
    unsigned long long mk = ((const unsigned long long*)(ws + WS_MK))[kk];
    int Qbit = (int)(mk >> 63) & 1;

    v8f Chi = {}, Clo = {}, Cpi = {};
    #pragma unroll
    for (int s = 0; s < 3; ++s) {
        Chi = __builtin_amdgcn_wmma_f32_16x16x32_f16(false, Ahi[s], false, Bv[s], (short)0, Chi, false, false);
        Clo = __builtin_amdgcn_wmma_f32_16x16x32_f16(false, Alo[s], false, Bv[s], (short)0, Clo, false, false);
        Cpi = __builtin_amdgcn_wmma_f32_16x16x32_f16(false, Api[s], false, Bv[s], (short)0, Cpi, false, false);
    }

    unsigned long long* cells = (unsigned long long*)(ws + WS_CELLS);
    int bloff = (lane >= 16) ? 8 : 0;                  // C layout: M = r + 8*(lane>=16)
    #pragma unroll
    for (int r = 0; r < 8; ++r) {
        int   bl = r + bloff;
        float D  = Chi[r] + Clo[r];                    // = sum d^2 (f32-accurate via hi/lo split)
        int psum = __float2int_rn(Cpi[r]);             // exact integer count of odd rints
        if ((psum + Qbit) & 1) {                       // D24 parity violated -> flip max coord
            float em = 0.0f;
            #pragma unroll
            for (int i = 0; i < DIM; ++i) {
                int m = (int)((mk >> (2 * i)) & 3ull);
                em = fmaxf(em, eT[bl * P96 + i * 4 + m]);
            }
            D += 1.0f - 2.0f * em;                     // (1-e)^2 - e^2
        }
        unsigned long long key =
            ((unsigned long long)__float_as_uint(D) << 32) | (unsigned long long)(unsigned)kk;
        atomicMin(&cells[brow * 16 + bl], key);        // bits of D>=0 are order-preserving
    }
}

// ---- kernel 4: recompute winning lattice point exactly -------------------
__global__ void leech_final(const float* __restrict__ x_in,
                            const int* __restrict__ C_rep,
                            const uint8_t* __restrict__ ws,
                            float* __restrict__ out)
{
    int b = blockIdx.x * blockDim.x + threadIdx.x;     // 0..1023
    unsigned long long cell = ((const unsigned long long*)(ws + WS_CELLS))[b];
    int k = (int)(unsigned)(cell & 0xFFFFFFFFull);
    const float* xr = x_in + b * DIM;
    const int*   cr = C_rep + k * DIM;

    float f[DIM], d[DIM];
    float fs = 0.0f;
    #pragma unroll
    for (int i = 0; i < DIM; ++i) {
        float t  = (xr[i] * INV_A - (float)cr[i]) * 0.25f;
        float fv = rintf(t);
        f[i] = fv; d[i] = t - fv; fs += fv;
    }
    if (((int)fs) & 1) {                               // parity fix: first argmax |d|
        float bm = -1.0f; int j = 0;
        #pragma unroll
        for (int i = 0; i < DIM; ++i) {
            float a = fabsf(d[i]);
            if (a > bm) { bm = a; j = i; }
        }
        float dj = d[j];
        f[j] += (dj > 0.0f) ? 1.0f : ((dj < 0.0f) ? -1.0f : 0.0f);  // round the other way
    }
    #pragma unroll
    for (int i = 0; i < DIM; ++i)
        out[b * DIM + i] = (4.0f * f[i] + (float)cr[i]) * A_SCALE;  // y = X * A
}

// ---- launcher ------------------------------------------------------------
extern "C" void kernel_launch(void* const* d_in, const int* in_sizes, int n_in,
                              void* d_out, int out_size, void* d_ws, size_t ws_size,
                              hipStream_t stream)
{
    (void)in_sizes; (void)n_in; (void)out_size; (void)ws_size;
    const float* x_in  = (const float*)d_in[0];
    const int*   C_rep = (const int*)d_in[1];
    float*       out   = (float*)d_out;
    uint8_t*     ws    = (uint8_t*)d_ws;

    leech_prep_tables<<<16, 128, 0, stream>>>(x_in, ws);          // 2048 threads
    leech_prep_cosets<<<32, 256, 0, stream>>>(C_rep, ws);         // 8192 threads
    leech_prep_bfrag <<<64, 256, 0, stream>>>(C_rep, ws);         // 16384 threads
    dim3 g(NBROW, NKTILE / 4);                                    // 64 x 128 blocks, 4 waves each
    leech_gemm<<<g, 128, 0, stream>>>(ws);
    leech_final<<<BATCH / 256, 256, 0, stream>>>(x_in, C_rep, ws, out);
}
